// UncertainGraphTraining_67224828117346
// MI455X (gfx1250) — matrix operand order repaired and checked by reading, not compiled, not run on hardware
//
#include <hip/hip_runtime.h>
#include <hip/hip_bf16.h>
#include <math.h>

// ---------------------------------------------------------------------------
// Sizes from the reference
// ---------------------------------------------------------------------------
#define N_NODES    50000
#define N_PAIRS    4096
#define N_EDGES    500000
#define N_SCORE    1000000
#define FEAT       300
#define HID        128
#define HEADS      3
#define OUTC       10
#define NEG_SLOPE  0.2f

typedef __attribute__((ext_vector_type(16))) _Float16     v16h;
typedef __attribute__((ext_vector_type(8)))  float        v8f;
typedef __attribute__((ext_vector_type(8)))  unsigned int v8u;
typedef __attribute__((ext_vector_type(4)))  float        f4;
typedef __attribute__((ext_vector_type(8)))  float        f8;
typedef __attribute__((ext_vector_type(8)))  _Float16     h8;

// ---------------------------------------------------------------------------
// WMMA GEMM:  C[M,N] = scale * (A[M,K] @ B) + bias
//   transB==0 : B row-major [K,N];  transB==1 : B row-major [N,K] (A @ B^T)
// Block = 256 threads = 8 waves (2 M x 4 N), block tile 64x64, K-step 32.
// Each wave computes 32x16 via two v_wmma_f32_16x16x32_f16 sharing one B frag.
// Interior tiles: vectorized branch-free staging (float4 loads, convertvector
// to fp16, ds_store_b128).  Edge tiles: clamped-address staging with
// multiply-masking so loads stay unconditional and batch under one wait.
// ---------------------------------------------------------------------------
#define BM 64
#define BN 64
#define BK 32
#define LDS_S 40   // halves per LDS row (pad: 80B stride, 16B aligned)

// ---- staging helpers -------------------------------------------------------
__device__ __forceinline__ void stage_a_fast(_Float16* __restrict__ As,
                                             const float* __restrict__ A,
                                             long blockM, int k0, int K, int t)
{
    int m  = t >> 2;           // 0..63
    int kq = (t & 3) << 3;     // 0,8,16,24
    const float* p = A + (blockM + m) * K + k0 + kq;
    f4 lo = *(const f4*)(p);
    f4 hi = *(const f4*)(p + 4);
    f8 f  = __builtin_shufflevector(lo, hi, 0, 1, 2, 3, 4, 5, 6, 7);
    *(h8*)(&As[m * LDS_S + kq]) = __builtin_convertvector(f, h8);
}

// clamped addresses (always-legal loads, issued unconditionally) +
// multiply-mask zeroing (keeps the load live so it cannot be predicated)
__device__ __forceinline__ void stage_a_checked(_Float16* __restrict__ As,
                                                const float* __restrict__ A,
                                                int blockM, int k0, int M, int K, int t)
{
    int m  = t >> 2;
    int kq = (t & 3) << 3;
    int gr = blockM + m;
    long rr = (long)min(gr, M - 1) * K;
    float rowm = (gr < M) ? 1.0f : 0.0f;
    float tmp[8];
#pragma unroll
    for (int j = 0; j < 8; ++j)
        tmp[j] = A[rr + min(k0 + kq + j, K - 1)];
    h8 h;
#pragma unroll
    for (int j = 0; j < 8; ++j) {
        float msk = rowm * ((k0 + kq + j < K) ? 1.0f : 0.0f);
        h[j] = (_Float16)(tmp[j] * msk);
    }
    *(h8*)(&As[m * LDS_S + kq]) = h;
}

// transB fast: contiguous along K (like A)
__device__ __forceinline__ void stage_bt_fast(_Float16* __restrict__ Bs,
                                              const float* __restrict__ B,
                                              long blockN, int k0, int K, int t)
{
    int n  = t >> 2;
    int kq = (t & 3) << 3;
    const float* p = B + (blockN + n) * K + k0 + kq;
    f4 lo = *(const f4*)(p);
    f4 hi = *(const f4*)(p + 4);
    f8 f  = __builtin_shufflevector(lo, hi, 0, 1, 2, 3, 4, 5, 6, 7);
    *(h8*)(&Bs[n * LDS_S + kq]) = __builtin_convertvector(f, h8);
}

// non-trans fast: contiguous along N, transposed into LDS [n][k]
__device__ __forceinline__ void stage_bn_fast(_Float16* __restrict__ Bs,
                                              const float* __restrict__ B,
                                              long blockN, int k0, int N, int t)
{
    int k  = t >> 3;            // 0..31
    int nq = (t & 7) << 3;      // 0..56
    const float* p = B + (long)(k0 + k) * N + blockN + nq;
    f4 lo = *(const f4*)(p);
    f4 hi = *(const f4*)(p + 4);
    f8 f  = __builtin_shufflevector(lo, hi, 0, 1, 2, 3, 4, 5, 6, 7);
    h8 h  = __builtin_convertvector(f, h8);
#pragma unroll
    for (int j = 0; j < 8; ++j) Bs[(nq + j) * LDS_S + k] = h[j];
}

// clamped + multiply-masked checked staging for B (both layouts)
__device__ __forceinline__ void stage_b_checked(_Float16* __restrict__ Bs,
                                                const float* __restrict__ B,
                                                int blockN, int k0, int N, int K,
                                                int transB, int t)
{
    int n  = t >> 2;
    int kq = (t & 3) << 3;
    int gn = blockN + n;
    int cn = min(gn, N - 1);
    float colm = (gn < N) ? 1.0f : 0.0f;
    float tmp[8];
    if (transB) {
        long rb = (long)cn * K;
#pragma unroll
        for (int j = 0; j < 8; ++j)
            tmp[j] = B[rb + min(k0 + kq + j, K - 1)];
    } else {
#pragma unroll
        for (int j = 0; j < 8; ++j)
            tmp[j] = B[(long)min(k0 + kq + j, K - 1) * N + cn];
    }
    h8 h;
#pragma unroll
    for (int j = 0; j < 8; ++j) {
        float msk = colm * ((k0 + kq + j < K) ? 1.0f : 0.0f);
        h[j] = (_Float16)(tmp[j] * msk);
    }
    *(h8*)(&Bs[n * LDS_S + kq]) = h;
}

// ---- fragment builders (CDNA5 16-bit WMMA VGPR layouts) --------------------
__device__ __forceinline__ v16h frag_a(const _Float16* __restrict__ row, int hi)
{
    v8u u;
#pragma unroll
    for (int p = 0; p < 8; ++p) {
        // A 16x32: lanes0-15 K pairs {0..7,16..23}; lanes16-31 shifted by 8
        int kb = 2 * p + ((p >= 4) ? 8 : 0) + (hi ? 8 : 0);
        u[p] = *(const unsigned int*)(row + kb);
    }
    return __builtin_bit_cast(v16h, u);
}
__device__ __forceinline__ v16h frag_b(const _Float16* __restrict__ row, int hi)
{
    v8u u;
#pragma unroll
    for (int p = 0; p < 8; ++p) {
        // B 32x16: lanes0-15 K pairs {0..7,16..23}; lanes16-31 shifted by 8
        int kb = 2 * p + ((p >= 4) ? 8 : 0) + (hi ? 8 : 0);
        u[p] = *(const unsigned int*)(row + kb);
    }
    return __builtin_bit_cast(v16h, u);
}

__device__ __forceinline__ void mma_step(const _Float16* __restrict__ As,
                                         const _Float16* __restrict__ Bs,
                                         int wm, int wn, int l16, int hi,
                                         v8f& acc0, v8f& acc1)
{
    v16h a0 = frag_a(&As[(wm * 32 + l16) * LDS_S], hi);
    v16h a1 = frag_a(&As[(wm * 32 + 16 + l16) * LDS_S], hi);
    v16h b  = frag_b(&Bs[(wn * 16 + l16) * LDS_S], hi);
    acc0 = __builtin_amdgcn_wmma_f32_16x16x32_f16(false, a0, false, b,
                                                  (short)0, acc0, false, false);
    acc1 = __builtin_amdgcn_wmma_f32_16x16x32_f16(false, a1, false, b,
                                                  (short)0, acc1, false, false);
}

__global__ __launch_bounds__(256)
void wmma_gemm_kernel(const float* __restrict__ A, const float* __restrict__ B,
                      const float* __restrict__ bias, float* __restrict__ C,
                      int M, int N, int K, int transB, float scale)
{
    __shared__ _Float16 As[BM * LDS_S];
    __shared__ _Float16 Bs[BN * LDS_S];

    const int t    = threadIdx.x;
    const int lane = t & 31;
    const int wave = t >> 5;
    const int wm   = wave >> 2;   // 0..1  -> rows wm*32 .. +31
    const int wn   = wave & 3;    // 0..3  -> cols wn*16 .. +15
    const int hi   = (lane >> 4) & 1;
    const int l16  = lane & 15;

    const int blockM = blockIdx.y * BM;
    const int blockN = blockIdx.x * BN;
    const bool fullM = (blockM + BM <= M);
    const bool fullN = (blockN + BN <= N);

    v8f acc0 = {}, acc1 = {};

    const int kfull = K & ~(BK - 1);
    for (int k0 = 0; k0 < kfull; k0 += BK) {
        if (fullM) stage_a_fast(As, A, blockM, k0, K, t);
        else       stage_a_checked(As, A, blockM, k0, M, K, t);
        if (fullN) {
            if (transB) stage_bt_fast(Bs, B, blockN, k0, K, t);
            else        stage_bn_fast(Bs, B, blockN, k0, N, t);
        } else {
            stage_b_checked(Bs, B, blockN, k0, N, K, transB, t);
        }
        if (k0 + BK < kfull) {
            // hint next K tile toward the caches (global_prefetch_b8)
            __builtin_prefetch(A + (long)(blockM + (t >> 2)) * K + k0 + BK, 0, 1);
        }
        __syncthreads();
        mma_step(As, Bs, wm, wn, l16, hi, acc0, acc1);
        __syncthreads();
    }
    if (kfull < K) {
        stage_a_checked(As, A, blockM, kfull, M, K, t);
        stage_b_checked(Bs, B, blockN, kfull, N, K, transB, t);
        __syncthreads();
        mma_step(As, Bs, wm, wn, l16, hi, acc0, acc1);
        __syncthreads();
    }

    // epilogue: C/D layout -> N = lane%16, element r -> M = r + 8*hi
    int col = blockN + wn * 16 + l16;
    if (col < N) {
        float bv = bias ? bias[col] : 0.0f;
#pragma unroll
        for (int r = 0; r < 8; ++r) {
            int row = blockM + wm * 32 + r + hi * 8;
            if (row < M) C[(long)row * N + col] = acc0[r] * scale + bv;
        }
#pragma unroll
        for (int r = 0; r < 8; ++r) {
            int row = blockM + wm * 32 + 16 + r + hi * 8;
            if (row < M) C[(long)row * N + col] = acc1[r] * scale + bv;
        }
    }
}

// ---------------------------------------------------------------------------
// Row softmax over [rows, ncols] in place (block per row)
// ---------------------------------------------------------------------------
__global__ __launch_bounds__(256)
void row_softmax_kernel(float* __restrict__ S, int ncols)
{
    __shared__ float red[256];
    float* r = S + (long)blockIdx.x * ncols;
    int t = threadIdx.x;

    float mx = -3.4e38f;
    for (int c = t; c < ncols; c += 256) mx = fmaxf(mx, r[c]);
    red[t] = mx; __syncthreads();
    for (int s = 128; s > 0; s >>= 1) {
        if (t < s) red[t] = fmaxf(red[t], red[t + s]);
        __syncthreads();
    }
    mx = red[0]; __syncthreads();

    float sum = 0.0f;
    for (int c = t; c < ncols; c += 256) {
        float e = __expf(r[c] - mx);
        r[c] = e; sum += e;
    }
    red[t] = sum; __syncthreads();
    for (int s = 128; s > 0; s >>= 1) {
        if (t < s) red[t] += red[t + s];
        __syncthreads();
    }
    float inv = 1.0f / red[0];
    for (int c = t; c < ncols; c += 256) r[c] *= inv;
}

// ---------------------------------------------------------------------------
// Elementwise helpers
// ---------------------------------------------------------------------------
__global__ void fill_f32_kernel(float* p, float v, long n) {
    long i = blockIdx.x * 256L + threadIdx.x;
    if (i < n) p[i] = v;
}
__global__ void fill_u32_kernel(unsigned int* p, unsigned int v, long n) {
    long i = blockIdx.x * 256L + threadIdx.x;
    if (i < n) p[i] = v;
}
__global__ void copy_f32_kernel(float* __restrict__ d, const float* __restrict__ s, long n) {
    long i = blockIdx.x * 256L + threadIdx.x;
    if (i < n) d[i] = s[i];
}
__global__ void vec_add_kernel(float* __restrict__ o, const float* __restrict__ a,
                               const float* __restrict__ b, long n) {
    long i = blockIdx.x * 256L + threadIdx.x;
    if (i < n) o[i] = a[i] + b[i];
}
__global__ void bias_elu_kernel(float* __restrict__ x, const float* __restrict__ bias,
                                int cols, long n) {
    long i = blockIdx.x * 256L + threadIdx.x;
    if (i >= n) return;
    float v = x[i] + bias[i % cols];
    x[i] = v > 0.0f ? v : (__expf(v) - 1.0f);
}
__global__ void add_bias_kernel(float* __restrict__ x, const float* __restrict__ bias,
                                int cols, long n) {
    long i = blockIdx.x * 256L + threadIdx.x;
    if (i >= n) return;
    x[i] = x[i] + bias[i % cols];
}

// Deterministic last-wins scatter: x_new[com_idx[p], :] = fused[p, :]
__global__ void scatter_rows_kernel(float* __restrict__ x_new,
                                    const float* __restrict__ fused,
                                    const int* __restrict__ com_idx,
                                    int npairs, int dim)
{
    int p = blockIdx.x * blockDim.x + threadIdx.x;
    if (p >= npairs) return;
    int idx = com_idx[p];
    for (int p2 = p + 1; p2 < npairs; ++p2)
        if (com_idx[p2] == idx) return;       // a later write wins
    const float* src = fused + (long)p * dim;
    float* dst = x_new + (long)idx * dim;
    for (int c = 0; c < dim; ++c) dst[c] = src[c];
}

// ---------------------------------------------------------------------------
// GAT attention machinery
// ---------------------------------------------------------------------------
__device__ __forceinline__ unsigned int f2ord(float f) {
    unsigned int u = __float_as_uint(f);
    return (u & 0x80000000u) ? ~u : (u | 0x80000000u);
}
__device__ __forceinline__ float ord2f(unsigned int u) {
    return (u & 0x80000000u) ? __uint_as_float(u ^ 0x80000000u)
                             : __uint_as_float(~u);
}

__global__ void node_dots_kernel(const float* __restrict__ h,
                                 const float* __restrict__ a_src,
                                 const float* __restrict__ a_dst,
                                 int n_nodes, int H, int C,
                                 float* __restrict__ esrc, float* __restrict__ edst)
{
    long i = blockIdx.x * 256L + threadIdx.x;
    if (i >= (long)n_nodes * H) return;
    int hh = (int)(i % H);
    const float* hp = h + i * C;               // contiguous [n][h][c]
    const float* as = a_src + hh * C;
    const float* ad = a_dst + hh * C;
    float s = 0.0f, d = 0.0f;
    for (int c = 0; c < C; ++c) { s += hp[c] * as[c]; d += hp[c] * ad[c]; }
    esrc[i] = s; edst[i] = d;
}

__global__ void edge_logits_kernel(const int* __restrict__ src, const int* __restrict__ dst,
                                   int E, int Etot, int H,
                                   const float* __restrict__ esrc,
                                   const float* __restrict__ edst,
                                   float* __restrict__ elog,
                                   unsigned int* __restrict__ m_ord)
{
    long i = blockIdx.x * 256L + threadIdx.x;
    if (i >= (long)Etot * H) return;
    int e = (int)(i / H), h = (int)(i % H);
    int s = (e < E) ? src[e] : (e - E);
    int d = (e < E) ? dst[e] : (e - E);
    float v = esrc[(long)s * H + h] + edst[(long)d * H + h];
    v = (v > 0.0f) ? v : NEG_SLOPE * v;
    elog[i] = v;
    atomicMax(&m_ord[(long)d * H + h], f2ord(v));
}

__global__ void finalize_max_kernel(unsigned int* __restrict__ m_ord, long n) {
    long i = blockIdx.x * 256L + threadIdx.x;
    if (i >= n) return;
    float v = ord2f(m_ord[i]);
    if (!isfinite(v)) v = 0.0f;
    ((float*)m_ord)[i] = v;                    // convert in place
}

__global__ void edge_exp_kernel(const int* __restrict__ dst,
                                int E, int Etot, int H,
                                const float* __restrict__ m,
                                float* __restrict__ elog,
                                float* __restrict__ denom)
{
    long i = blockIdx.x * 256L + threadIdx.x;
    if (i >= (long)Etot * H) return;
    int e = (int)(i / H), h = (int)(i % H);
    int d = (e < E) ? dst[e] : (e - E);
    float ex = __expf(elog[i] - m[(long)d * H + h]);
    elog[i] = ex;
    atomicAdd(&denom[(long)d * H + h], ex);
}

__global__ void edge_agg_kernel(const int* __restrict__ src, const int* __restrict__ dst,
                                int E, int Etot, int H, int C,
                                const float* __restrict__ hfeat,
                                const float* __restrict__ elog,
                                const float* __restrict__ denom,
                                float* __restrict__ out)
{
    long i = blockIdx.x * 256L + threadIdx.x;
    if (i >= (long)Etot * H) return;
    int e = (int)(i / H), h = (int)(i % H);
    int s = (e < E) ? src[e] : (e - E);
    int d = (e < E) ? dst[e] : (e - E);
    float alpha = elog[i] / (denom[(long)d * H + h] + 1e-16f);
    const float* hs = hfeat + ((long)s * H + h) * C;
    float* od = out + ((long)d * H + h) * C;
    for (int c = 0; c < C; ++c) atomicAdd(&od[c], hs[c] * alpha);
}

__global__ void pair_scores_kernel(const float* __restrict__ emb,
                                   const int* __restrict__ pair_idx,
                                   float* __restrict__ out, int npairs, int C)
{
    long p = blockIdx.x * 256L + threadIdx.x;
    if (p >= npairs) return;
    int a = pair_idx[2 * p], b = pair_idx[2 * p + 1];
    const float* ea = emb + (long)a * C;
    const float* eb = emb + (long)b * C;
    float s = 0.0f;
    for (int c = 0; c < C; ++c) s += ea[c] * eb[c];
    out[p] = s;
}

// ---------------------------------------------------------------------------
// Launch helpers
// ---------------------------------------------------------------------------
static inline dim3 gemm_grid(int M, int N) {
    return dim3((N + BN - 1) / BN, (M + BM - 1) / BM);
}
static inline long nblk(long n) { return (n + 255) / 256; }

extern "C" void kernel_launch(void* const* d_in, const int* in_sizes, int n_in,
                              void* d_out, int out_size, void* d_ws, size_t ws_size,
                              hipStream_t stream)
{
    const float* fsub   = (const float*)d_in[0];
    const float* fcom   = (const float*)d_in[1];
    const float* x      = (const float*)d_in[2];
    const int*   comidx = (const int*)  d_in[3];
    const int*   eidx   = (const int*)  d_in[4];
    const int*   pidx   = (const int*)  d_in[5];
    const float* Wq = (const float*)d_in[6];  const float* bq = (const float*)d_in[7];
    const float* Wk = (const float*)d_in[8];  const float* bk = (const float*)d_in[9];
    const float* Wv = (const float*)d_in[10]; const float* bv = (const float*)d_in[11];
    const float* Wf = (const float*)d_in[12]; const float* bf = (const float*)d_in[13];
    const float* W1 = (const float*)d_in[14];
    const float* as1 = (const float*)d_in[15];
    const float* ad1 = (const float*)d_in[16];
    const float* b1  = (const float*)d_in[17];
    const float* W2 = (const float*)d_in[18];
    const float* as2 = (const float*)d_in[19];
    const float* ad2 = (const float*)d_in[20];
    const float* b2  = (const float*)d_in[21];
    float* out = (float*)d_out;

    const int* esrc_idx = eidx;                 // row 0
    const int* edst_idx = eidx + N_EDGES;       // row 1
    const int  ET = N_EDGES + N_NODES;          // edges + self loops

    // ---------------- workspace layout (floats) ----------------
    float* w = (float*)d_ws;
    size_t o = 0;
    float* q      = w + o; o += (size_t)N_PAIRS * FEAT;
    float* kmat   = w + o; o += (size_t)N_PAIRS * FEAT;
    float* vmat   = w + o; o += (size_t)N_PAIRS * FEAT;
    float* att    = w + o; o += (size_t)N_PAIRS * FEAT;
    float* fused  = w + o; o += (size_t)N_PAIRS * FEAT;
    float* scores = w + o; o += (size_t)N_PAIRS * N_PAIRS;
    float* x_new  = w + o; o += (size_t)N_NODES * FEAT;
    float* h1     = w + o; o += (size_t)N_NODES * HEADS * HID;
    float* out1   = w + o; o += (size_t)N_NODES * HEADS * HID;
    float* es1    = w + o; o += (size_t)N_NODES * HEADS;
    float* ed1    = w + o; o += (size_t)N_NODES * HEADS;
    float* elog1  = w + o; o += (size_t)ET * HEADS;
    float* m1     = w + o; o += (size_t)N_NODES * HEADS;   // ord u32 then float
    float* den1   = w + o; o += (size_t)N_NODES * HEADS;
    float* h2     = w + o; o += (size_t)N_NODES * OUTC;
    float* out2   = w + o; o += (size_t)N_NODES * OUTC;
    float* es2    = w + o; o += (size_t)N_NODES;
    float* ed2    = w + o; o += (size_t)N_NODES;
    float* elog2  = w + o; o += (size_t)ET;
    float* m2     = w + o; o += (size_t)N_NODES;
    float* den2   = w + o; o += (size_t)N_NODES;
    (void)ws_size; (void)n_in; (void)in_sizes; (void)out_size;

    const float inv_sqrt_d = 1.0f / sqrtf((float)FEAT);

    // ---------------- cross-attention ----------------
    wmma_gemm_kernel<<<gemm_grid(N_PAIRS, FEAT), 256, 0, stream>>>(
        fsub, Wq, bq, q, N_PAIRS, FEAT, FEAT, 0, 1.0f);
    wmma_gemm_kernel<<<gemm_grid(N_PAIRS, FEAT), 256, 0, stream>>>(
        fcom, Wk, bk, kmat, N_PAIRS, FEAT, FEAT, 0, 1.0f);
    wmma_gemm_kernel<<<gemm_grid(N_PAIRS, FEAT), 256, 0, stream>>>(
        fcom, Wv, bv, vmat, N_PAIRS, FEAT, FEAT, 0, 1.0f);
    // scores = (q @ k^T) / sqrt(d)
    wmma_gemm_kernel<<<gemm_grid(N_PAIRS, N_PAIRS), 256, 0, stream>>>(
        q, kmat, nullptr, scores, N_PAIRS, N_PAIRS, FEAT, 1, inv_sqrt_d);
    row_softmax_kernel<<<N_PAIRS, 256, 0, stream>>>(scores, N_PAIRS);
    // att = scores @ v
    wmma_gemm_kernel<<<gemm_grid(N_PAIRS, FEAT), 256, 0, stream>>>(
        scores, vmat, nullptr, att, N_PAIRS, FEAT, N_PAIRS, 0, 1.0f);
    // q buffer is dead: reuse for (fsub + att)
    vec_add_kernel<<<nblk((long)N_PAIRS * FEAT), 256, 0, stream>>>(
        q, fsub, att, (long)N_PAIRS * FEAT);
    wmma_gemm_kernel<<<gemm_grid(N_PAIRS, FEAT), 256, 0, stream>>>(
        q, Wf, bf, fused, N_PAIRS, FEAT, FEAT, 0, 1.0f);

    // ---------------- write fused rows into node features ----------------
    copy_f32_kernel<<<nblk((long)N_NODES * FEAT), 256, 0, stream>>>(
        x_new, x, (long)N_NODES * FEAT);
    scatter_rows_kernel<<<nblk(N_PAIRS), 256, 0, stream>>>(
        x_new, fused, comidx, N_PAIRS, FEAT);

    // ---------------- GAT layer 1 ----------------
    wmma_gemm_kernel<<<gemm_grid(N_NODES, HEADS * HID), 256, 0, stream>>>(
        x_new, W1, nullptr, h1, N_NODES, HEADS * HID, FEAT, 0, 1.0f);
    node_dots_kernel<<<nblk((long)N_NODES * HEADS), 256, 0, stream>>>(
        h1, as1, ad1, N_NODES, HEADS, HID, es1, ed1);

    fill_u32_kernel<<<nblk((long)N_NODES * HEADS), 256, 0, stream>>>(
        (unsigned int*)m1, 0x007FFFFFu /* f2ord(-inf) */, (long)N_NODES * HEADS);
    fill_f32_kernel<<<nblk((long)N_NODES * HEADS), 256, 0, stream>>>(
        den1, 0.0f, (long)N_NODES * HEADS);
    fill_f32_kernel<<<nblk((long)N_NODES * HEADS * HID), 256, 0, stream>>>(
        out1, 0.0f, (long)N_NODES * HEADS * HID);

    edge_logits_kernel<<<nblk((long)ET * HEADS), 256, 0, stream>>>(
        esrc_idx, edst_idx, N_EDGES, ET, HEADS, es1, ed1, elog1, (unsigned int*)m1);
    finalize_max_kernel<<<nblk((long)N_NODES * HEADS), 256, 0, stream>>>(
        (unsigned int*)m1, (long)N_NODES * HEADS);
    edge_exp_kernel<<<nblk((long)ET * HEADS), 256, 0, stream>>>(
        edst_idx, N_EDGES, ET, HEADS, m1, elog1, den1);
    edge_agg_kernel<<<nblk((long)ET * HEADS), 256, 0, stream>>>(
        esrc_idx, edst_idx, N_EDGES, ET, HEADS, HID, h1, elog1, den1, out1);
    bias_elu_kernel<<<nblk((long)N_NODES * HEADS * HID), 256, 0, stream>>>(
        out1, b1, HEADS * HID, (long)N_NODES * HEADS * HID);

    // ---------------- GAT layer 2 ----------------
    wmma_gemm_kernel<<<gemm_grid(N_NODES, OUTC), 256, 0, stream>>>(
        out1, W2, nullptr, h2, N_NODES, OUTC, HEADS * HID, 0, 1.0f);
    node_dots_kernel<<<nblk((long)N_NODES), 256, 0, stream>>>(
        h2, as2, ad2, N_NODES, 1, OUTC, es2, ed2);

    fill_u32_kernel<<<nblk((long)N_NODES), 256, 0, stream>>>(
        (unsigned int*)m2, 0x007FFFFFu, (long)N_NODES);
    fill_f32_kernel<<<nblk((long)N_NODES), 256, 0, stream>>>(den2, 0.0f, (long)N_NODES);
    fill_f32_kernel<<<nblk((long)N_NODES * OUTC), 256, 0, stream>>>(
        out2, 0.0f, (long)N_NODES * OUTC);

    edge_logits_kernel<<<nblk((long)ET), 256, 0, stream>>>(
        esrc_idx, edst_idx, N_EDGES, ET, 1, es2, ed2, elog2, (unsigned int*)m2);
    finalize_max_kernel<<<nblk((long)N_NODES), 256, 0, stream>>>(
        (unsigned int*)m2, (long)N_NODES);
    edge_exp_kernel<<<nblk((long)ET), 256, 0, stream>>>(
        edst_idx, N_EDGES, ET, 1, m2, elog2, den2);
    edge_agg_kernel<<<nblk((long)ET), 256, 0, stream>>>(
        esrc_idx, edst_idx, N_EDGES, ET, 1, OUTC, h2, elog2, den2, out2);
    add_bias_kernel<<<nblk((long)N_NODES * OUTC), 256, 0, stream>>>(
        out2, b2, OUTC, (long)N_NODES * OUTC);

    // ---------------- pair link scores ----------------
    pair_scores_kernel<<<nblk((long)N_SCORE), 256, 0, stream>>>(
        out2, pidx, out, N_SCORE, OUTC);
}
